// MultiHead_24885040513047
// MI455X (gfx1250) — matrix operand order repaired and checked by reading, not compiled, not run on hardware
//
#include <hip/hip_runtime.h>
#include <math.h>

// ---------------------------------------------------------------------------
// Problem constants (B=2, S=2048, D=512, H=8, HD=64)
// ---------------------------------------------------------------------------
#define BB   2
#define SS   2048
#define DD   512
#define HH   8
#define HD   64
#define MROWS (BB * SS)            // 4096 flattened rows
#define LN_EPS 1e-5f

typedef __attribute__((ext_vector_type(16))) __bf16 v16bf;
typedef __attribute__((ext_vector_type(8)))  __bf16 v8bf;   // 16 bytes
typedef __attribute__((ext_vector_type(8)))  float  v8f;
typedef __attribute__((ext_vector_type(4)))  float  f4;

static __device__ __forceinline__ v16bf cat8(v8bf lo, v8bf hi) {
    return __builtin_shufflevector(lo, hi, 0, 1, 2, 3, 4, 5, 6, 7,
                                           8, 9, 10, 11, 12, 13, 14, 15);
}

// A-fragment (16x32, bf16 row-major source): per lane row fixed by caller,
// K elements {16g..16g+15}.
static __device__ __forceinline__ v16bf load_a_bf(const __bf16* __restrict__ rowp,
                                                  int g) {
    const __bf16* ap = rowp + 16 * g;
    return cat8(*(const v8bf*)ap, *(const v8bf*)(ap + 8));
}

// B-fragment (32x16) where B[k][n] is contiguous in k at p (k = 16g base):
// b[i] = p[i], i = 0..15.
static __device__ __forceinline__ v16bf load_b_bf(const __bf16* __restrict__ p) {
    return cat8(*(const v8bf*)p, *(const v8bf*)(p + 8));
}

// ---------------------------------------------------------------------------
// Kernel 0: fp32 -> bf16 bulk convert (n multiple of 8)
// ---------------------------------------------------------------------------
__global__ __launch_bounds__(256) void cvt_bf16_kernel(
    const float* __restrict__ src, __bf16* __restrict__ dst, int n8)
{
    const int idx = blockIdx.x * blockDim.x + threadIdx.x;
    if (idx >= n8) return;
    const int base = idx * 8;
    f4 a = *(const f4*)(src + base);
    f4 b = *(const f4*)(src + base + 4);
    v8bf o;
    o[0] = (__bf16)a[0]; o[1] = (__bf16)a[1]; o[2] = (__bf16)a[2]; o[3] = (__bf16)a[3];
    o[4] = (__bf16)b[0]; o[5] = (__bf16)b[1]; o[6] = (__bf16)b[2]; o[7] = (__bf16)b[3];
    *(v8bf*)(dst + base) = o;
}

// ---------------------------------------------------------------------------
// Kernel 1: fused QKV projection  y = x @ W^T + bias   (all-bf16 operands)
// grid (32, 8, 3), block 256 (8 waves). Wave -> 16x64 tile.
// Q, K written bf16 row-major [4096, 512]; V written bf16 transposed [B, D, S].
// ---------------------------------------------------------------------------
__global__ __launch_bounds__(256) void qkv_proj_kernel(
    const __bf16* __restrict__ xq, const __bf16* __restrict__ xk,
    const __bf16* __restrict__ xv,
    const __bf16* __restrict__ wq, const __bf16* __restrict__ wk,
    const __bf16* __restrict__ wv,
    const float* __restrict__ bq, const float* __restrict__ bk,
    const float* __restrict__ bv,
    __bf16* __restrict__ Qo, __bf16* __restrict__ Ko, __bf16* __restrict__ Vt)
{
    const int z = blockIdx.z;
    const __bf16* x    = (z == 0) ? xq : (z == 1) ? xk : xv;
    const __bf16* w    = (z == 0) ? wq : (z == 1) ? wk : wv;
    const float*  bias = (z == 0) ? bq : (z == 1) ? bk : bv;

    const int lane = threadIdx.x & 31;
    const int wave = threadIdx.x >> 5;
    const int g    = lane >> 4;
    const int ln   = lane & 15;

    const int m0 = (blockIdx.x * 8 + wave) * 16;
    const int n0 = blockIdx.y * 64;

    v8f acc[4] = {};
    const int arow = m0 + ln;

    for (int k0 = 0; k0 < DD; k0 += 32) {
        v16bf a = load_a_bf(x + (size_t)arow * DD + k0, g);
#pragma unroll
        for (int c = 0; c < 4; ++c) {
            const int col = n0 + c * 16 + ln;
            v16bf bfr = load_b_bf(w + (size_t)col * DD + k0 + 16 * g);
            acc[c] = __builtin_amdgcn_wmma_f32_16x16x32_bf16(
                false, a, false, bfr, (short)0, acc[c], false, false);
        }
    }
#pragma unroll
    for (int c = 0; c < 4; ++c) {
        const int col = n0 + c * 16 + ln;
        const float bvv = bias[col];
#pragma unroll
        for (int r = 0; r < 8; ++r) {
            const int row = m0 + r + 8 * g;
            const __bf16 val = (__bf16)(acc[c][r] + bvv);
            if (z == 0) {
                Qo[(size_t)row * DD + col] = val;
            } else if (z == 1) {
                Ko[(size_t)row * DD + col] = val;
            } else {
                const int bb = row >> 11;          // row / S
                const int sr = row & (SS - 1);     // row % S
                Vt[((size_t)(bb * DD + col)) * SS + sr] = val;
            }
        }
    }
}

// ---------------------------------------------------------------------------
// Kernel 2: causal flash attention. grid (32, B*H=16), block 128 (4 waves),
// wave -> one 16-query tile. All matrix operands bf16, stats fp32.
// ---------------------------------------------------------------------------
__global__ __launch_bounds__(128) void attn_kernel(
    const __bf16* __restrict__ Q, const __bf16* __restrict__ K,
    const __bf16* __restrict__ Vt, __bf16* __restrict__ CTX)
{
    const int wave = threadIdx.x >> 5;
    const int lane = threadIdx.x & 31;
    const int g    = lane >> 4;
    const int ln   = lane & 15;

    const int qt = blockIdx.x * 4 + wave;         // 0..127
    const int bh = blockIdx.y;
    const int b  = bh >> 3;
    const int h  = bh & 7;
    const int q0 = qt * 16;
    const size_t base  = (size_t)b * SS * DD + (size_t)h * HD;   // Q/K/CTX base
    const size_t vbase = ((size_t)(b * DD + h * HD)) * SS;       // Vt base

    __shared__ __bf16 pst[4][16 * 32];            // per-wave P staging
    __bf16* my = pst[wave];

    // Q fragments (two 32-wide head-dim chunks)
    v16bf qa[2];
#pragma unroll
    for (int s = 0; s < 2; ++s)
        qa[s] = load_a_bf(Q + base + (size_t)(q0 + ln) * DD + s * 32, g);

    float m_r[8], l_r[8];
#pragma unroll
    for (int r = 0; r < 8; ++r) { m_r[r] = -INFINITY; l_r[r] = 0.0f; }
    v8f octx[4] = {};

    const float invsc = 1.0f / (8.0f + 1e-8f);    // 1/(sqrt(64)+1e-8)
    const int kend = q0 + 16;

    for (int k0 = 0; k0 < kend; k0 += 32) {
        // ---- scores: 16x32 = Q(16x64) x K^T(64x32), two 16-key subtiles ----
        v8f sc[2] = {};
#pragma unroll
        for (int t = 0; t < 2; ++t) {
            const int key = k0 + t * 16 + ln;
#pragma unroll
            for (int s = 0; s < 2; ++s) {
                v16bf kb = load_b_bf(K + base + (size_t)key * DD + s * 32 + 16 * g);
                sc[t] = __builtin_amdgcn_wmma_f32_16x16x32_bf16(
                    false, qa[s], false, kb, (short)0, sc[t], false, false);
            }
        }
        // ---- scale + causal mask (row = q0+r+8g, col = k0+t*16+ln) ----
#pragma unroll
        for (int t = 0; t < 2; ++t) {
            const int col = k0 + t * 16 + ln;
#pragma unroll
            for (int r = 0; r < 8; ++r) {
                const int row = q0 + r + 8 * g;
                float s = sc[t][r] * invsc;
                if (col > row) s = -INFINITY;
                sc[t][r] = s;
            }
        }
        // ---- online softmax stats ----
        float nm[8], alpha[8];
#pragma unroll
        for (int r = 0; r < 8; ++r) {
            float mx = fmaxf(sc[0][r], sc[1][r]);
#pragma unroll
            for (int off = 1; off < 16; off <<= 1)
                mx = fmaxf(mx, __shfl_xor(mx, off, 32));
            nm[r]    = fmaxf(m_r[r], mx);
            alpha[r] = __expf(m_r[r] - nm[r]);
            m_r[r]   = nm[r];
        }
#pragma unroll
        for (int r = 0; r < 8; ++r) {
            float p0 = __expf(sc[0][r] - nm[r]);
            float p1 = __expf(sc[1][r] - nm[r]);
            sc[0][r] = p0; sc[1][r] = p1;
            float s = p0 + p1;
#pragma unroll
            for (int off = 1; off < 16; off <<= 1)
                s += __shfl_xor(s, off, 32);
            l_r[r] = l_r[r] * alpha[r] + s;
        }
#pragma unroll
        for (int c = 0; c < 4; ++c)
#pragma unroll
            for (int r = 0; r < 8; ++r) octx[c][r] *= alpha[r];

        // ---- relayout P: C-layout regs -> LDS -> A-layout fragment ----
#pragma unroll
        for (int t = 0; t < 2; ++t)
#pragma unroll
            for (int r = 0; r < 8; ++r)
                my[(r + 8 * g) * 32 + t * 16 + ln] = (__bf16)sc[t][r];
        asm volatile("s_wait_dscnt 0" ::: "memory");
        const __bf16* prow = my + ln * 32 + 16 * g;
        v16bf pa = cat8(*(const v8bf*)prow, *(const v8bf*)(prow + 8));
        asm volatile("s_wait_dscnt 0" ::: "memory");

        // ---- ctx += P(16x32) x V(32x64): Vt gives contiguous B-frags ----
#pragma unroll
        for (int c = 0; c < 4; ++c) {
            v16bf vb = load_b_bf(Vt + vbase + (size_t)(c * 16 + ln) * SS
                                 + k0 + 16 * g);
            octx[c] = __builtin_amdgcn_wmma_f32_16x16x32_bf16(
                false, pa, false, vb, (short)0, octx[c], false, false);
        }
    }

    // ---- normalize and write bf16 context (merged-head layout [B,S,D]) ----
    float inv[8];
#pragma unroll
    for (int r = 0; r < 8; ++r) inv[r] = 1.0f / l_r[r];
#pragma unroll
    for (int c = 0; c < 4; ++c)
#pragma unroll
        for (int r = 0; r < 8; ++r) {
            const int row = q0 + r + 8 * g;
            const int col = c * 16 + ln;
            CTX[base + (size_t)row * DD + col] = (__bf16)(octx[c][r] * inv[r]);
        }
}

// ---------------------------------------------------------------------------
// Kernel 3: o = ctx @ wo^T + bo ; x = residual + o ; LayerNorm(x)
// grid 256, block 256 (8 waves) -> one block owns a full 16x512 row strip.
// ---------------------------------------------------------------------------
__global__ __launch_bounds__(256) void out_ln_kernel(
    const __bf16* __restrict__ CTX, const __bf16* __restrict__ WO,
    const float* __restrict__ BO,  const float* __restrict__ RES,
    const float* __restrict__ gamma, const float* __restrict__ beta,
    float* __restrict__ out)
{
    const int lane = threadIdx.x & 31;
    const int wave = threadIdx.x >> 5;
    const int g    = lane >> 4;
    const int ln   = lane & 15;

    const int m0 = blockIdx.x * 16;
    const int n0 = wave * 64;

    v8f acc[4] = {};
    const int arow = m0 + ln;

    for (int k0 = 0; k0 < DD; k0 += 32) {
        v16bf a = load_a_bf(CTX + (size_t)arow * DD + k0, g);
#pragma unroll
        for (int c = 0; c < 4; ++c) {
            const int col = n0 + c * 16 + ln;
            v16bf bfr = load_b_bf(WO + (size_t)col * DD + k0 + 16 * g);
            acc[c] = __builtin_amdgcn_wmma_f32_16x16x32_bf16(
                false, a, false, bfr, (short)0, acc[c], false, false);
        }
    }

    // x = gemm + bias + residual
    float xv[4][8];
#pragma unroll
    for (int c = 0; c < 4; ++c) {
        const int col = n0 + c * 16 + ln;
        const float bvv = BO[col];
#pragma unroll
        for (int r = 0; r < 8; ++r) {
            const int row = m0 + r + 8 * g;
            xv[c][r] = acc[c][r] + bvv + RES[(size_t)row * DD + col];
        }
    }

    // block-wide mean / var per row (16 rows x 512 cols)
    __shared__ float ssum[8][16];
    __shared__ float ssq[8][16];
#pragma unroll
    for (int r = 0; r < 8; ++r) {
        float s = xv[0][r] + xv[1][r] + xv[2][r] + xv[3][r];
        float q = xv[0][r] * xv[0][r] + xv[1][r] * xv[1][r] +
                  xv[2][r] * xv[2][r] + xv[3][r] * xv[3][r];
#pragma unroll
        for (int off = 1; off < 16; off <<= 1) {
            s += __shfl_xor(s, off, 32);
            q += __shfl_xor(q, off, 32);
        }
        if (ln == 0) {
            ssum[wave][r + 8 * g] = s;
            ssq [wave][r + 8 * g] = q;
        }
    }
    __syncthreads();

#pragma unroll
    for (int r = 0; r < 8; ++r) {
        const int row = r + 8 * g;
        float tot = 0.0f, totq = 0.0f;
#pragma unroll
        for (int w = 0; w < 8; ++w) { tot += ssum[w][row]; totq += ssq[w][row]; }
        const float mu  = tot * (1.0f / (float)DD);
        const float var = totq * (1.0f / (float)DD) - mu * mu;
        const float rs  = rsqrtf(var + LN_EPS);
        const int grow  = m0 + row;
#pragma unroll
        for (int c = 0; c < 4; ++c) {
            const int col = n0 + c * 16 + ln;
            out[(size_t)grow * DD + col] =
                gamma[col] * (xv[c][r] - mu) * rs + beta[col];
        }
    }
}

// ---------------------------------------------------------------------------
// Launch
// ---------------------------------------------------------------------------
extern "C" void kernel_launch(void* const* d_in, const int* in_sizes, int n_in,
                              void* d_out, int out_size, void* d_ws, size_t ws_size,
                              hipStream_t stream) {
    const float* q     = (const float*)d_in[0];
    const float* k     = (const float*)d_in[1];
    const float* v     = (const float*)d_in[2];
    const float* wq    = (const float*)d_in[3];
    const float* bq    = (const float*)d_in[4];
    const float* wk    = (const float*)d_in[5];
    const float* bk    = (const float*)d_in[6];
    const float* wv    = (const float*)d_in[7];
    const float* bv    = (const float*)d_in[8];
    const float* wo    = (const float*)d_in[9];
    const float* bo    = (const float*)d_in[10];
    const float* gamma = (const float*)d_in[11];
    const float* beta  = (const float*)d_in[12];
    // d_in[13] is the causal mask — applied analytically in attn_kernel.

    float* out = (float*)d_out;
    __bf16* wsb = (__bf16*)d_ws;

    const size_t NA = (size_t)MROWS * DD;   // 2M: activation tensor
    const size_t NW = (size_t)DD * DD;      // 256K: weight matrix

    __bf16* xq_b = wsb;                 // 3 converted inputs
    __bf16* xk_b = xq_b + NA;
    __bf16* xv_b = xk_b + NA;
    __bf16* wq_b = xv_b + NA;           // 4 converted weights
    __bf16* wk_b = wq_b + NW;
    __bf16* wv_b = wk_b + NW;
    __bf16* wo_b = wv_b + NW;
    __bf16* Qb   = wo_b + NW;           // projected Q (row-major)
    __bf16* Kb   = Qb + NA;             // projected K (row-major)
    __bf16* Vtb  = Kb + NA;             // projected V (transposed [B,D,S])
    __bf16* Cb   = Vtb + NA;            // attention context (row-major)

    // --- stage 0: bulk fp32 -> bf16 conversion ---
    {
        const int na8 = (int)(NA / 8), nw8 = (int)(NW / 8);
        dim3 blk(256);
        dim3 ga((na8 + 255) / 256), gw((nw8 + 255) / 256);
        cvt_bf16_kernel<<<ga, blk, 0, stream>>>(q,  xq_b, na8);
        cvt_bf16_kernel<<<ga, blk, 0, stream>>>(k,  xk_b, na8);
        cvt_bf16_kernel<<<ga, blk, 0, stream>>>(v,  xv_b, na8);
        cvt_bf16_kernel<<<gw, blk, 0, stream>>>(wq, wq_b, nw8);
        cvt_bf16_kernel<<<gw, blk, 0, stream>>>(wk, wk_b, nw8);
        cvt_bf16_kernel<<<gw, blk, 0, stream>>>(wv, wv_b, nw8);
        cvt_bf16_kernel<<<gw, blk, 0, stream>>>(wo, wo_b, nw8);
    }

    dim3 g1(32, 8, 3), b1(256);
    qkv_proj_kernel<<<g1, b1, 0, stream>>>(xq_b, xk_b, xv_b, wq_b, wk_b, wv_b,
                                           bq, bk, bv, Qb, Kb, Vtb);

    dim3 g2(32, BB * HH), b2(128);
    attn_kernel<<<g2, b2, 0, stream>>>(Qb, Kb, Vtb, Cb);

    dim3 g3(MROWS / 16), b3(256);
    out_ln_kernel<<<g3, b3, 0, stream>>>(Cb, wo_b, bo, q, gamma, beta, out);
}
